// AttentionHead_69174743270123
// MI455X (gfx1250) — compile-verified, matrix operand order checked by reading, not run on hardware
//
#include <hip/hip_runtime.h>
#include <hip/hip_bf16.h>

typedef __attribute__((ext_vector_type(16))) __bf16 v16bf;
typedef __attribute__((ext_vector_type(8)))  float  v8f;
typedef __attribute__((ext_vector_type(4)))  int    v4i;

union FragAB {
  v16bf v;
  uint4 q[2];
};

#define TILE_M 128
#define TILE_N 128
#define TILE_K 32
#define LDS_ROW 40   // 32 bf16 + 8 pad (80B rows -> conflict-light b128 frag reads)

// ---- CDNA5 async global->LDS path (guarded; sync fallback always compiles) --
#if defined(__HIP_DEVICE_COMPILE__) && __has_builtin(__builtin_amdgcn_global_load_async_to_lds_b128)
#define HAVE_ASYNC_LDS 1
#else
#define HAVE_ASYNC_LDS 0
#endif

__device__ __forceinline__ void wait_async_lds() {
#if HAVE_ASYNC_LDS
#if __has_builtin(__builtin_amdgcn_s_wait_asynccnt)
  __builtin_amdgcn_s_wait_asynccnt(0);
#else
  asm volatile("s_wait_asynccnt 0x0" ::: "memory");
#endif
#endif
}

__device__ __forceinline__ void copy16_g2l(const __bf16* g, __bf16* l) {
#if HAVE_ASYNC_LDS
  __builtin_amdgcn_global_load_async_to_lds_b128(
      (__attribute__((address_space(1))) v4i*)(g),
      (__attribute__((address_space(3))) v4i*)(l), 0, 0);
#else
  *(uint4*)l = *(const uint4*)g;
#endif
}

// ---------------------------------------------------------------------------
// Tiled bf16 GEMM, TN form: C[m][n] = sum_k A[m][k] * Bt[n][k]  (+ bias[n])
// OUT_MODE: 0 = f32 row-major, 1 = f32 transposed (C[n][m]),
//           2 = bf16 row-major, 3 = bf16 transposed
// Grid: (N/128, M/128, batch); sAb/sBb/sCb are per-batch element strides.
// Double-buffered LDS; async global->LDS staging when available.
// ---------------------------------------------------------------------------
template<int OUT_MODE>
__global__ __launch_bounds__(256)
void gemm_bf16_tn(const __bf16* __restrict__ A, long long sAb,
                  const __bf16* __restrict__ Bt, long long sBb,
                  void* __restrict__ C, long long sCb,
                  const float* __restrict__ bias,
                  int K, int lda, int ldb, int ldc)
{
  __shared__ __align__(16) __bf16 sA[2][TILE_M][LDS_ROW];
  __shared__ __align__(16) __bf16 sB[2][TILE_N][LDS_ROW];

  const int tid  = threadIdx.x;
  const int lane = tid & 31;
  const int wave = tid >> 5;
  const int wm   = wave >> 1;   // 0..3 : 32-row band of the 128x128 tile
  const int wn   = wave & 1;    // 0..1 : 64-col band
  const int half = lane >> 4;   // 0/1  : lane group (ISA 16-bit WMMA layout)
  const int r    = lane & 15;

  const int m0 = blockIdx.y * TILE_M;
  const int n0 = blockIdx.x * TILE_N;

  A  += (long long)blockIdx.z * sAb;
  Bt += (long long)blockIdx.z * sBb;

  v8f zero = {};
  v8f acc[2][4];
#pragma unroll
  for (int i = 0; i < 2; ++i)
#pragma unroll
    for (int j = 0; j < 4; ++j) acc[i][j] = zero;

  // staging coords: 512 uint4 chunks per 128x32 tile, 2 per thread per matrix
  const int c0 = tid, c1 = tid + 256;
  const int ra0 = c0 >> 2, ca0 = c0 & 3;
  const int ra1 = c1 >> 2, ca1 = c1 & 3;
  const __bf16* gA0 = A  + (long long)(m0 + ra0) * lda + ca0 * 8;
  const __bf16* gA1 = A  + (long long)(m0 + ra1) * lda + ca1 * 8;
  const __bf16* gB0 = Bt + (long long)(n0 + ra0) * ldb + ca0 * 8;
  const __bf16* gB1 = Bt + (long long)(n0 + ra1) * ldb + ca1 * 8;

  auto stage = [&](int buf, int kt) {
    copy16_g2l(gA0 + kt, &sA[buf][ra0][ca0 * 8]);
    copy16_g2l(gA1 + kt, &sA[buf][ra1][ca1 * 8]);
    copy16_g2l(gB0 + kt, &sB[buf][ra0][ca0 * 8]);
    copy16_g2l(gB1 + kt, &sB[buf][ra1][ca1 * 8]);
  };

  stage(0, 0);
  wait_async_lds();
  __syncthreads();

  int cur = 0;
  for (int kt = 0; kt < K; kt += TILE_K) {
    // prefetch next K-tile into the other buffer (overlaps the WMMAs below)
    if (kt + TILE_K < K) stage(cur ^ 1, kt + TILE_K);

    // ---- fragments (16-bit A/B layout: e0..7 -> K=half*8+0..7,
    //      e8..15 -> K=16+half*8+0..7; two b128 LDS reads per fragment) ----
    const __bf16 (*tA)[LDS_ROW] = sA[cur];
    const __bf16 (*tB)[LDS_ROW] = sB[cur];
    FragAB a[2], b[4];
#pragma unroll
    for (int tm = 0; tm < 2; ++tm) {
      const int row = wm * 32 + tm * 16 + r;
      a[tm].q[0] = *(const uint4*)&tA[row][half * 8];
      a[tm].q[1] = *(const uint4*)&tA[row][16 + half * 8];
    }
#pragma unroll
    for (int tn = 0; tn < 4; ++tn) {
      const int col = wn * 64 + tn * 16 + r;
      b[tn].q[0] = *(const uint4*)&tB[col][half * 8];
      b[tn].q[1] = *(const uint4*)&tB[col][16 + half * 8];
    }

#pragma unroll
    for (int tm = 0; tm < 2; ++tm)
#pragma unroll
      for (int tn = 0; tn < 4; ++tn)
        acc[tm][tn] = __builtin_amdgcn_wmma_f32_16x16x32_bf16(
            false, a[tm].v, false, b[tn].v, (short)0, acc[tm][tn],
            false, false);

    wait_async_lds();     // producer LDS writes landed
    __syncthreads();      // lowers to s_wait_dscnt 0 + barrier: reads done too
    cur ^= 1;
  }

  // ---- epilogue: D layout is VGPR i -> m = i + 8*half, n = lane%16 ----
  const long long cbase = (long long)blockIdx.z * sCb;
#pragma unroll
  for (int tm = 0; tm < 2; ++tm) {
#pragma unroll
    for (int tn = 0; tn < 4; ++tn) {
      const int nn  = n0 + wn * 64 + tn * 16 + r;
      const float bvn = bias ? bias[nn] : 0.0f;
#pragma unroll
      for (int i = 0; i < 8; ++i) {
        const int mm = m0 + wm * 32 + tm * 16 + half * 8 + i;
        const float val = acc[tm][tn][i] + bvn;
        if (OUT_MODE == 0)
          ((float*)C)[cbase + (long long)mm * ldc + nn] = val;
        else if (OUT_MODE == 1)
          ((float*)C)[cbase + (long long)nn * ldc + mm] = val;
        else if (OUT_MODE == 2)
          ((__bf16*)C)[cbase + (long long)mm * ldc + nn] = (__bf16)val;
        else
          ((__bf16*)C)[cbase + (long long)nn * ldc + mm] = (__bf16)val;
      }
    }
  }
}

// ---------------------------------------------------------------------------
__global__ void cvt_f32_to_bf16(const float* __restrict__ x,
                                __bf16* __restrict__ y, int n)
{
  int i = blockIdx.x * blockDim.x + threadIdx.x;
  const int stride = gridDim.x * blockDim.x;
  for (; i < n; i += stride) y[i] = (__bf16)x[i];
}

__global__ void cvt_transpose_f32_to_bf16(const float* __restrict__ W,
                                          __bf16* __restrict__ Wt,
                                          int D, int H)
{
  const int i = blockIdx.x * blockDim.x + threadIdx.x;   // over D*H
  if (i >= D * H) return;
  const int k = i / H, n = i - k * H;
  Wt[(long long)n * D + k] = (__bf16)W[i];
}

// ---------------------------------------------------------------------------
// Row softmax over St[b][g][0..H-1] (softmax over h per column g of scores),
// scale 1/sqrt(1024), mask[b][g]==0 -> -inf row (NaN result, as in reference).
// Writes bf16 probabilities IN PLACE over the f32 row (first 2KB of the 4KB
// row); consumer GEMM uses ldb = 2*H bf16 elements to stride rows.
// ---------------------------------------------------------------------------
__global__ __launch_bounds__(256)
void softmax_row_inplace(float* __restrict__ St, const int* __restrict__ mask,
                         int G, int H)
{
  __shared__ float red[8];
  const int row = blockIdx.x;          // b*G + g
  const int b = row / G, g = row - b * G;
  float* x = St + (long long)row * H;
  const bool live = mask[b * G + g] != 0;
  const float scale = 0.03125f;        // 1/sqrt(1024)

  float vals[4];
  float mx = -3.4e38f;
#pragma unroll
  for (int t = 0; t < 4; ++t) {
    float v = x[threadIdx.x + 256 * t];
    v = live ? v * scale : -__builtin_inff();
    vals[t] = v;
    mx = fmaxf(mx, v);
  }
#pragma unroll
  for (int off = 16; off > 0; off >>= 1) mx = fmaxf(mx, __shfl_xor(mx, off, 32));
  if ((threadIdx.x & 31) == 0) red[threadIdx.x >> 5] = mx;
  __syncthreads();
  mx = red[0];
#pragma unroll
  for (int w = 1; w < 8; ++w) mx = fmaxf(mx, red[w]);
  __syncthreads();

  float e[4];
  float s = 0.f;
#pragma unroll
  for (int t = 0; t < 4; ++t) { e[t] = __expf(vals[t] - mx); s += e[t]; }
#pragma unroll
  for (int off = 16; off > 0; off >>= 1) s += __shfl_xor(s, off, 32);
  if ((threadIdx.x & 31) == 0) red[threadIdx.x >> 5] = s;
  __syncthreads();
  s = 0.f;
#pragma unroll
  for (int w = 0; w < 8; ++w) s += red[w];
  const float inv = 1.0f / s;

  __syncthreads();                      // all f32 reads done before aliasing
  __bf16* y = (__bf16*)x;
#pragma unroll
  for (int t = 0; t < 4; ++t)
    y[threadIdx.x + 256 * t] = (__bf16)(e[t] * inv);
}

// ---------------------------------------------------------------------------
extern "C" void kernel_launch(void* const* d_in, const int* in_sizes, int n_in,
                              void* d_out, int out_size, void* d_ws, size_t ws_size,
                              hipStream_t stream)
{
  (void)in_sizes; (void)n_in; (void)out_size; (void)ws_size;
  const float* hidden = (const float*)d_in[0];
  const int*   amask  = (const int*)d_in[1];
  const float* wq = (const float*)d_in[2];
  const float* bq = (const float*)d_in[3];
  const float* wk = (const float*)d_in[4];
  const float* bk = (const float*)d_in[5];
  const float* wv = (const float*)d_in[6];
  const float* bv = (const float*)d_in[7];
  float* out = (float*)d_out;

  const int Bn = 16, S = 1024, Dm = 1024, Hh = 1024;
  const long long BS = (long long)Bn * S;    // 16384 rows total

  char* ws = (char*)d_ws;
  size_t off = 0;
  __bf16* Xbf = (__bf16*)(ws + off); off += (size_t)BS * Dm * 2;        // 32MB
  __bf16* Wqt = (__bf16*)(ws + off); off += (size_t)Dm * Hh * 2;        //  2MB
  __bf16* Wkt = (__bf16*)(ws + off); off += (size_t)Dm * Hh * 2;
  __bf16* Wvt = (__bf16*)(ws + off); off += (size_t)Dm * Hh * 2;
  __bf16* Qt  = (__bf16*)(ws + off); off += (size_t)BS * Hh * 2;        // [b][h][s]
  __bf16* Kt  = (__bf16*)(ws + off); off += (size_t)BS * Hh * 2;        // [b][h][s]
  __bf16* Vbf = (__bf16*)(ws + off); off += (size_t)BS * Hh * 2;        // [b][s][h]
  float*  St  = (float*)(ws + off);                                     // [b][g][h]

  // 1. precision conversion (one-time, bandwidth-trivial)
  cvt_f32_to_bf16<<<4096, 256, 0, stream>>>(hidden, Xbf, (int)(BS * Dm));
  cvt_transpose_f32_to_bf16<<<(Dm * Hh) / 256, 256, 0, stream>>>(wq, Wqt, Dm, Hh);
  cvt_transpose_f32_to_bf16<<<(Dm * Hh) / 256, 256, 0, stream>>>(wk, Wkt, Dm, Hh);
  cvt_transpose_f32_to_bf16<<<(Dm * Hh) / 256, 256, 0, stream>>>(wv, Wvt, Dm, Hh);

  // 2. QKV projections (batched over b; weights shared, stride 0)
  //    Q,K stored transposed ([b][h][s]) for K-contiguous scores operands.
  dim3 gproj(Hh / TILE_N, S / TILE_M, Bn);
  gemm_bf16_tn<3><<<gproj, 256, 0, stream>>>(
      Xbf, (long long)S * Dm, Wqt, 0, Qt, (long long)Hh * S, bq, Dm, Dm, Dm, S);
  gemm_bf16_tn<3><<<gproj, 256, 0, stream>>>(
      Xbf, (long long)S * Dm, Wkt, 0, Kt, (long long)Hh * S, bk, Dm, Dm, Dm, S);
  gemm_bf16_tn<2><<<gproj, 256, 0, stream>>>(
      Xbf, (long long)S * Dm, Wvt, 0, Vbf, (long long)S * Hh, bv, Dm, Dm, Dm, Hh);

  // 3. scores[h][g] = sum_s Kt[h][s]*Qt[g][s], stored transposed: St[g][h]
  dim3 gsc(Hh / TILE_N, Hh / TILE_M, Bn);
  gemm_bf16_tn<1><<<gsc, 256, 0, stream>>>(
      Kt, (long long)Hh * S, Qt, (long long)Hh * S, St, (long long)Hh * Hh,
      nullptr, S, S, S, Hh);

  // 4. softmax over h (contiguous rows of St); bf16 probs in place
  softmax_row_inplace<<<Bn * Hh, 256, 0, stream>>>(St, amask, Hh, Hh);

  // 5. out[s][g] = sum_h V[s][h] * P[g][h]  (P rows alias St rows, ldb=2H)
  dim3 gout(Hh / TILE_N, S / TILE_M, Bn);
  gemm_bf16_tn<0><<<gout, 256, 0, stream>>>(
      Vbf, (long long)S * Hh, (const __bf16*)St, (long long)Hh * 2048,
      out, (long long)S * Hh, nullptr, Hh, Hh, 2048, Hh);
}
